// GeneratorWithBN_89867895702145
// MI455X (gfx1250) — matrix operand order repaired and checked by reading, not compiled, not run on hardware
//
#include <hip/hip_runtime.h>
#include <hip/hip_bf16.h>
#include <cstdint>
#include <cstddef>

// ---------------------------------------------------------------------------
// GeneratorWithBN for MI455X (gfx1250, wave32, WMMA + TDM).
// B=16384, DIN=1, H=512, DOUT=256. Two GEMMs via v_wmma_f32_16x16x32_f16;
// B-tile staged into LDS by the Tensor Data Mover (tensor_load_to_lds with
// LDS row padding for bank-conflict-free ds_load_b128). All BN reductions
// are fixed-order trees (deterministic). Workspace use: ~51.2 MB.
// ---------------------------------------------------------------------------

#define BATCH 16384
#define HDIM  512
#define ODIM  256
#define NNODE 64
#define EPSV  1e-5f
#define KDIM  512          // K of both GEMMs (compile-time)
#define LDSK  520          // 512 + 8 f16 pad -> row stride 1040B = 260 dw = 4 mod 64

typedef _Float16 half8  __attribute__((ext_vector_type(8)));
typedef _Float16 half16 __attribute__((ext_vector_type(16)));
typedef float    float8 __attribute__((ext_vector_type(8)));
typedef unsigned int u32x4 __attribute__((ext_vector_type(4)));
typedef int          i32x4 __attribute__((ext_vector_type(4)));
typedef int          i32x8 __attribute__((ext_vector_type(8)));

#if defined(__has_builtin)
#  if __has_builtin(__builtin_amdgcn_tensor_load_to_lds)
#    define HAVE_TDM 1
#  else
#    define HAVE_TDM 0
#  endif
#else
#  define HAVE_TDM 0
#endif

__device__ __forceinline__ float leaky02(float v) { return fmaxf(v, 0.2f * v); }
__device__ __forceinline__ float sigm(float v)    { return 1.0f / (1.0f + __expf(-v)); }

// ---- 1. scalar batch stats of x -------------------------------------------
__global__ __launch_bounds__(256)
void k_stats_x(const float* __restrict__ x, float* __restrict__ stats, int n) {
  __shared__ float s1[256], s2[256];
  const int t = threadIdx.x;
  float a = 0.f, b = 0.f;
  for (int i = t; i < n; i += 256) { float v = x[i]; a += v; b += v * v; }
  s1[t] = a; s2[t] = b;
  __syncthreads();
  for (int s = 128; s > 0; s >>= 1) {
    if (t < s) { s1[t] += s1[t + s]; s2[t] += s2[t + s]; }
    __syncthreads();
  }
  if (t == 0) {
    float m = s1[0] / n;
    stats[0] = m;
    stats[1] = s2[0] / n - m * m;   // population variance (ddof=0)
  }
}

// ---- 2. per-column coefficients -------------------------------------------
// Rank-1 BN identity: BN(x*w)[b,h] = (x[b]-mx) * w*g*rsqrt(vx*w^2+eps) + be.
// Node update (DIN=1 layernorm collapse): col <- alpha + beta*col.
__global__ __launch_bounds__(512)
void k_prep(const float* __restrict__ W0,  const float* __restrict__ g0,
            const float* __restrict__ Ws0, const float* __restrict__ gs0,
            const float* __restrict__ Ws1, const float* __restrict__ gs1,
            const int*   __restrict__ node_idx,
            const float* __restrict__ tbn2, const float* __restrict__ cbn,
            const float* __restrict__ stats,
            float* __restrict__ A0, float* __restrict__ S0, float* __restrict__ S1,
            float* __restrict__ colA, float* __restrict__ colB) {
  const int h = threadIdx.x;
  const float vx = stats[1];
  if (h < HDIM) {
    float w  = W0[h];  A0[h] = w  * g0[h]  * rsqrtf(vx * w  * w  + EPSV);
    float w1 = Ws0[h]; S0[h] = w1 * gs0[h] * rsqrtf(vx * w1 * w1 + EPSV);
    float w2 = Ws1[h]; S1[h] = w2 * gs1[h] * rsqrtf(vx * w2 * w2 + EPSV);
    colA[h] = 0.f; colB[h] = 1.f;
  }
  __syncthreads();
  if (h < NNODE) {
    int c = node_idx[h];
    float g = sigm(tbn2[h]);            // pi == tbn2 (LN over size-1 axis)
    colA[c] = g * fmaxf(cbn[h], 0.f);   // g * relu(upd), upd == cbn
    colB[c] = 1.f - g;
  }
}

// ---- 3. f32 -> f16 weight conversion --------------------------------------
__global__ __launch_bounds__(256)
void k_cvt_f16(const float* __restrict__ src, _Float16* __restrict__ dst, int n) {
  int i = blockIdx.x * 256 + threadIdx.x;
  if (i < n) dst[i] = (_Float16)src[i];
}

// ---- 4. fused layer-0 -> f16 A --------------------------------------------
__global__ __launch_bounds__(256)
void k_layer0(const float* __restrict__ x,
              const float* __restrict__ be0, const float* __restrict__ bes0,
              const float* __restrict__ stats,
              const float* __restrict__ A0, const float* __restrict__ S0,
              const float* __restrict__ colA, const float* __restrict__ colB,
              _Float16* __restrict__ Aout) {
  __shared__ float red[256];
  const int b = blockIdx.x, t = threadIdx.x;
  const float xc = x[b] - stats[0];
  const int h0 = t, h1 = t + 256;
  float m0 = leaky02(xc * A0[h0] + be0[h0]);
  float m1 = leaky02(xc * A0[h1] + be0[h1]);
  red[t] = m0 + m1;
  __syncthreads();
  for (int s = 128; s > 0; s >>= 1) { if (t < s) red[t] += red[t + s]; __syncthreads(); }
  const float g = sigm(red[0] * (1.0f / HDIM));
  float s0 = xc * S0[h0] + bes0[h0];
  float s1 = xc * S0[h1] + bes0[h1];
  float v0 = g * m0 + (1.f - g) * s0;  v0 = colA[h0] + colB[h0] * v0;
  float v1 = g * m1 + (1.f - g) * s1;  v1 = colA[h1] + colB[h1] * v1;
  Aout[(size_t)b * HDIM + h0] = (_Float16)v0;
  Aout[(size_t)b * HDIM + h1] = (_Float16)v1;
}

// ---- 5. WMMA GEMM: C[M,N] = A[M,KDIM](f16 rm) * W[N,KDIM]^T ----------------
// Block: 8 waves, 128(M)x64(N); wave: 16(M)x64(N) = 4 accumulators.
// B tile (64 rows x K, 64KB) staged to LDS once per block by the TDM.
// K-loop fully unrolled (compile-time K): straight-line load/wmma stream.
template <int N>
__global__ __launch_bounds__(256)
void k_gemm_f16(const _Float16* __restrict__ A, const _Float16* __restrict__ W,
                float* __restrict__ C) {
  extern __shared__ _Float16 Bs[];          // 64 * LDSK f16 = 66560 B dynamic LDS
  const int wave = threadIdx.x >> 5;
  const int lane = threadIdx.x & 31;
  const int hl   = lane >> 4;               // lane half: selects K sub-range
  const int l16  = lane & 15;
  const int m0   = blockIdx.x * 128 + wave * 16;
  const int nB   = blockIdx.y * 64;

#if HAVE_TDM
  if (threadIdx.x < 32) {                   // wave 0 issues the DMA (EXEC ignored)
    const uint64_t gaddr = (uint64_t)(uintptr_t)(W + (size_t)nB * KDIM);
    const uint32_t laddr = (uint32_t)(uintptr_t)&Bs[0];   // flat addr[31:0] == LDS offset
    u32x4 g0; i32x8 g1; i32x4 z4 = {};
    // D# group0: count=1, lds_addr, global_addr[56:0], type=2
    g0[0] = 1u;
    g0[1] = laddr;
    g0[2] = (uint32_t)gaddr;
    g0[3] = (uint32_t)((gaddr >> 32) & 0x1FFFFFFu) | (2u << 30);
    // D# group1: data_size=2B, pad 4 dw every 256 dw; 2-D tile 512 x 64
    g1[0] = (1 << 16) | (1 << 20) | (7 << 22) | (3 << 25);
    g1[1] = (int)(512u << 16);              // tensor_dim0 = 512 (lo 16 in bits 31:16)
    g1[2] = (int)(64u << 16);               // tensor_dim0 hi=0 | tensor_dim1 = 64
    g1[3] = (int)(512u << 16);              // tensor_dim1 hi=0 | tile_dim0 = 512
    g1[4] = 64;                             // tile_dim1 = 64, tile_dim2 = 0
    g1[5] = 512;                            // tensor_dim0_stride = 512
    g1[6] = 0; g1[7] = 0;
#if __clang_major__ >= 23
    { i32x8 z8 = {}; __builtin_amdgcn_tensor_load_to_lds(g0, g1, z4, z4, z8, 0); }
#else
    __builtin_amdgcn_tensor_load_to_lds(g0, g1, z4, z4, 0);
#endif
    __builtin_amdgcn_s_wait_tensorcnt(0);
  }
#else
  for (int i = threadIdx.x; i < 64 * (KDIM / 8); i += 256) {
    int r = i >> 6, c = (i & 63) * 8;       // 64 x (K/8) 16B chunks
    *reinterpret_cast<half8*>(&Bs[r * LDSK + c]) =
        *reinterpret_cast<const half8*>(&W[(size_t)(nB + r) * KDIM + c]);
  }
#endif
  __syncthreads();

  const _Float16* Arow = A + (size_t)(m0 + l16) * KDIM;
  float8 acc[4] = {};
  // Fully unrolled K loop, two independent A frags per body (no carried copy).
#pragma unroll
  for (int kk = 0; kk < KDIM; kk += 64) {
    if (kk + 64 < KDIM)
      __builtin_prefetch(Arow + kk + 64, 0, 3);   // global_prefetch_b8
    // A 16x32 f16 frags: lanes 0-15 hold K {kk..+15},
    //                    lanes 16-31 hold K {kk+16..+31}
    union AF { half16 v; half8 h[2]; } af0, af1;
    af0.h[0] = *reinterpret_cast<const half8*>(Arow + kk + hl * 16);
    af0.h[1] = *reinterpret_cast<const half8*>(Arow + kk + 8 + hl * 16);
    af1.h[0] = *reinterpret_cast<const half8*>(Arow + kk + 32 + hl * 16);
    af1.h[1] = *reinterpret_cast<const half8*>(Arow + kk + 40 + hl * 16);
#pragma unroll
    for (int u = 0; u < 2; ++u) {
      const int k0 = kk + u * 32;
      const half16 av = (u == 0) ? af0.v : af1.v;
      // Load all 4 B frags from LDS first (clause), then 4 back-to-back WMMAs.
      half16 bf0 = *reinterpret_cast<const half16*>(&Bs[(0 * 16 + l16) * LDSK + k0 + hl * 16]);
      half16 bf1 = *reinterpret_cast<const half16*>(&Bs[(1 * 16 + l16) * LDSK + k0 + hl * 16]);
      half16 bf2 = *reinterpret_cast<const half16*>(&Bs[(2 * 16 + l16) * LDSK + k0 + hl * 16]);
      half16 bf3 = *reinterpret_cast<const half16*>(&Bs[(3 * 16 + l16) * LDSK + k0 + hl * 16]);
      acc[0] = __builtin_amdgcn_wmma_f32_16x16x32_f16(false, av, false, bf0, (short)0, acc[0], false, false);
      acc[1] = __builtin_amdgcn_wmma_f32_16x16x32_f16(false, av, false, bf1, (short)0, acc[1], false, false);
      acc[2] = __builtin_amdgcn_wmma_f32_16x16x32_f16(false, av, false, bf2, (short)0, acc[2], false, false);
      acc[3] = __builtin_amdgcn_wmma_f32_16x16x32_f16(false, av, false, bf3, (short)0, acc[3], false, false);
    }
  }
#pragma unroll
  for (int nt = 0; nt < 4; ++nt) {
    const int col = nB + nt * 16 + l16;
#pragma unroll
    for (int r = 0; r < 8; ++r)   // C/D: VGPR r -> M = r + 8*hl, lane%16 -> N
      C[(size_t)(m0 + r + 8 * hl) * N + col] = acc[nt][r];
  }
}

// ---- 6. per-column BN stats (deterministic tree) ---------------------------
__global__ __launch_bounds__(256)
void k_colstats(const float* __restrict__ C, int ldc, int rows,
                float* __restrict__ cmean, float* __restrict__ crstd) {
  __shared__ float s1[256], s2[256];
  const int j = blockIdx.x, t = threadIdx.x;
  float a = 0.f, b = 0.f;
  for (int r = t; r < rows; r += 256) {
    float v = C[(size_t)r * ldc + j]; a += v; b += v * v;
  }
  s1[t] = a; s2[t] = b;
  __syncthreads();
  for (int s = 128; s > 0; s >>= 1) {
    if (t < s) { s1[t] += s1[t + s]; s2[t] += s2[t + s]; }
    __syncthreads();
  }
  if (t == 0) {
    float m = s1[0] / rows;
    cmean[j] = m;
    crstd[j] = rsqrtf(s2[0] / rows - m * m + EPSV);
  }
}

// ---- 7. fused layer-1 elementwise -> f16 -----------------------------------
__global__ __launch_bounds__(256)
void k_layer1(const float* __restrict__ x, const float* __restrict__ C,
              const float* __restrict__ cm, const float* __restrict__ cr,
              const float* __restrict__ g1, const float* __restrict__ be1,
              const float* __restrict__ bes1, const float* __restrict__ stats,
              const float* __restrict__ S1,
              const float* __restrict__ colA, const float* __restrict__ colB,
              _Float16* __restrict__ Aout) {
  __shared__ float red[256];
  const int b = blockIdx.x, t = threadIdx.x;
  const float xc = x[b] - stats[0];
  const int h0 = t, h1 = t + 256;
  const size_t base = (size_t)b * HDIM;
  float m0 = leaky02((C[base + h0] - cm[h0]) * cr[h0] * g1[h0] + be1[h0]);
  float m1 = leaky02((C[base + h1] - cm[h1]) * cr[h1] * g1[h1] + be1[h1]);
  red[t] = m0 + m1;
  __syncthreads();
  for (int s = 128; s > 0; s >>= 1) { if (t < s) red[t] += red[t + s]; __syncthreads(); }
  const float g = sigm(red[0] * (1.0f / HDIM));
  float s0 = xc * S1[h0] + bes1[h0];
  float s1v = xc * S1[h1] + bes1[h1];
  float v0 = g * m0 + (1.f - g) * s0;   v0 = colA[h0] + colB[h0] * v0;
  float v1 = g * m1 + (1.f - g) * s1v;  v1 = colA[h1] + colB[h1] * v1;
  Aout[base + h0] = (_Float16)v0;
  Aout[base + h1] = (_Float16)v1;
}

// ---- 8. output: tanh(BN) ---------------------------------------------------
__global__ __launch_bounds__(256)
void k_out(const float* __restrict__ C, const float* __restrict__ cm,
           const float* __restrict__ cr, const float* __restrict__ go,
           const float* __restrict__ beo, float* __restrict__ out) {
  const int b = blockIdx.x, o = threadIdx.x;
  const size_t i = (size_t)b * ODIM + o;
  out[i] = tanhf((C[i] - cm[o]) * cr[o] * go[o] + beo[o]);
}

// ---------------------------------------------------------------------------
extern "C" void kernel_launch(void* const* d_in, const int* in_sizes, int n_in,
                              void* d_out, int out_size, void* d_ws, size_t ws_size,
                              hipStream_t stream) {
  (void)in_sizes; (void)n_in; (void)out_size; (void)ws_size;
  // input order == setup_inputs() dict order (params dict insertion order)
  const float* x    = (const float*)d_in[0];
  const float* W0   = (const float*)d_in[1];
  const float* g0   = (const float*)d_in[3];
  const float* be0  = (const float*)d_in[4];
  const float* Ws0  = (const float*)d_in[5];
  const float* gs0  = (const float*)d_in[7];
  const float* bes0 = (const float*)d_in[8];
  const float* W1   = (const float*)d_in[9];
  const float* g1   = (const float*)d_in[11];
  const float* be1  = (const float*)d_in[12];
  const float* Ws1  = (const float*)d_in[13];
  const float* gs1  = (const float*)d_in[15];
  const float* bes1 = (const float*)d_in[16];
  const float* Wo   = (const float*)d_in[17];
  const float* go   = (const float*)d_in[19];
  const float* beo  = (const float*)d_in[20];
  const float* tbn2 = (const float*)d_in[34];
  const float* cbn  = (const float*)d_in[38];
  const int*   nidx = (const int*)  d_in[40];
  // (biases b*/bs*/bo and all attention/parent params cancel mathematically)

  char* ws = (char*)d_ws;
  float*     Cbuf = (float*)ws;                                  // 32 MB
  _Float16*  Abuf = (_Float16*)(ws + 33554432);                  // 16 MB
  _Float16*  W1h  = (_Float16*)(ws + 50331648);                  // 512 KB
  _Float16*  Woh  = (_Float16*)(ws + 50855936);                  // 256 KB
  float* sm   = (float*)(ws + 51118080);
  float* stats = sm;           // 2
  float* A0    = sm + 16;      // 512
  float* S0    = A0 + HDIM;    // 512
  float* S1    = S0 + HDIM;    // 512
  float* colA  = S1 + HDIM;    // 512
  float* colB  = colA + HDIM;  // 512
  float* cm1   = colB + HDIM;  // 512
  float* cr1   = cm1 + HDIM;   // 512
  float* cm2   = cr1 + HDIM;   // 256
  float* cr2   = cm2 + ODIM;   // 256

  const size_t smemB = 64 * LDSK * sizeof(_Float16);             // 66560 B

  k_stats_x<<<1, 256, 0, stream>>>(x, stats, BATCH);
  k_prep<<<1, 512, 0, stream>>>(W0, g0, Ws0, gs0, Ws1, gs1, nidx, tbn2, cbn,
                                stats, A0, S0, S1, colA, colB);
  k_cvt_f16<<<(HDIM * HDIM) / 256, 256, 0, stream>>>(W1, W1h, HDIM * HDIM);
  k_cvt_f16<<<(ODIM * HDIM) / 256, 256, 0, stream>>>(Wo, Woh, ODIM * HDIM);

  k_layer0<<<BATCH, 256, 0, stream>>>(x, be0, bes0, stats, A0, S0, colA, colB, Abuf);
  k_gemm_f16<HDIM><<<dim3(BATCH / 128, HDIM / 64), 256, smemB, stream>>>(Abuf, W1h, Cbuf);
  k_colstats<<<HDIM, 256, 0, stream>>>(Cbuf, HDIM, BATCH, cm1, cr1);
  k_layer1<<<BATCH, 256, 0, stream>>>(x, Cbuf, cm1, cr1, g1, be1, bes1, stats,
                                      S1, colA, colB, Abuf);
  k_gemm_f16<ODIM><<<dim3(BATCH / 128, ODIM / 64), 256, smemB, stream>>>(Abuf, Woh, Cbuf);
  k_colstats<<<ODIM, 256, 0, stream>>>(Cbuf, ODIM, BATCH, cm2, cr2);
  k_out<<<BATCH, 256, 0, stream>>>(Cbuf, cm2, cr2, go, beo, (float*)d_out);
}
